// IDS4Block_58926951301790
// MI455X (gfx1250) — matrix operand order repaired and checked by reading, not compiled, not run on hardware
//
#include <hip/hip_runtime.h>

// ---------------------------------------------------------------------------
// IDS4Block for MI455X (gfx1250), wave32 + WMMA + async-to-LDS staging.
//   K1: A_ = gelu(x @ A)            bf16 WMMA GEMM  (8192x1024)@(1024x4096)
//   K2: per-chunk (G=64) running product (f32 WMMA, LDS-resident) with
//       double-buffered GLOBAL_LOAD_ASYNC_TO_LDS_B128 staging of A_t;
//       emits r_t = Lloc_t @ proj_w and chunk totals T_c
//   K3: per-batch scan of chunk totals -> offsets E_c (f32 WMMA, async-staged)
//   K4: s_t = E_c @ r_t + proj_b    (f32 WMMA, 16 positions per wave)
//   K5: out = s @ C^T + x @ D^T + Cb + Db   bf16 WMMA GEMM
// ---------------------------------------------------------------------------

typedef __attribute__((ext_vector_type(2)))  float  v2f;
typedef __attribute__((ext_vector_type(8)))  float  v8f;
typedef __attribute__((ext_vector_type(4)))  int    v4i;
typedef __attribute__((address_space(3)))    v4i    lds_v4i;
typedef __attribute__((ext_vector_type(16))) __bf16 v16bf;

#define B_   4
#define L_   2048
#define DM_  1024
#define DS_  64
#define BL_  (B_*L_)          // 8192 rows
#define NN_  (DS_*DS_)        // 4096
#define G_   64               // scan chunk length
#define NC_  (L_/G_)          // 32 chunks per batch
#define LDST 66               // padded LDS row stride for the running product

#if __has_builtin(__builtin_amdgcn_global_load_async_to_lds_b128) && \
    __has_builtin(__builtin_amdgcn_s_wait_asynccnt)
#define HAS_ASYNC 1
#else
#define HAS_ASYNC 0
#endif

__device__ __forceinline__ float gelu_exact(float v) {
  return 0.5f * v * (1.0f + erff(v * 0.70710678118654752f));
}

// Stage one contiguous 64x64 f32 tile (16 KB) global -> LDS.
// 512 threads x 2 b128 ops; per-wave ASYNCcnt += 2.
__device__ __forceinline__ void tile_load_async(float* dst, const float* src, int tid) {
#if HAS_ASYNC
  __builtin_amdgcn_global_load_async_to_lds_b128(
      (v4i*)(uintptr_t)(src + tid * 4),
      (lds_v4i*)(uintptr_t)(dst + tid * 4), 0, 0);
  __builtin_amdgcn_global_load_async_to_lds_b128(
      (v4i*)(uintptr_t)(src + 2048 + tid * 4),
      (lds_v4i*)(uintptr_t)(dst + 2048 + tid * 4), 0, 0);
#else
  ((float4*)dst)[tid]       = ((const float4*)src)[tid];
  ((float4*)dst)[tid + 512] = ((const float4*)src)[tid + 512];
#endif
}

template <int N>
__device__ __forceinline__ void tile_wait() {
#if HAS_ASYNC
  __builtin_amdgcn_s_wait_asynccnt(N);      // immediate operand
#endif
}

// ------------------------- K1: A_ = gelu(x @ A) ----------------------------
__global__ __launch_bounds__(256)
void k1_xA_gelu(const float* __restrict__ x, const float* __restrict__ A,
                float* __restrict__ Ag) {
  const int NTN = NN_ / 16;                    // 256 tiles in N
  int gw   = (int)((blockIdx.x * blockDim.x + threadIdx.x) >> 5);
  int lane = threadIdx.x & 31;
  int tm = gw / NTN, tn = gw % NTN;
  int hi = (lane >= 16) ? 1 : 0;
  int mr = lane & 15;
  size_t m = (size_t)tm * 16 + mr;
  size_t n = (size_t)tn * 16 + mr;
  v8f c = {};
  for (int k0 = 0; k0 < DM_; k0 += 32) {
    v16bf a, b;
    const float* xr = x + m * DM_ + k0 + (hi ? 16 : 0);
    const float* wc = A + (size_t)(k0 + (hi ? 16 : 0)) * NN_ + n;
    __builtin_prefetch(xr + 32, 0, 1);         // global_prefetch_b8
    #pragma unroll
    for (int i = 0; i < 16; ++i) a[i] = (__bf16)xr[i];
    #pragma unroll
    for (int i = 0; i < 16; ++i) b[i] = (__bf16)wc[(size_t)i * NN_];
    c = __builtin_amdgcn_wmma_f32_16x16x32_bf16(false, a, false, b,
                                                (short)0, c, false, false);
  }
  int rb = tm * 16 + (hi ? 8 : 0);
  #pragma unroll
  for (int i = 0; i < 8; ++i)
    Ag[(size_t)(rb + i) * NN_ + n] = gelu_exact(c[i]);
}

// ---------------- K2: chunk-local prefix products (FP32 WMMA) --------------
__global__ __launch_bounds__(512)
void k2_chunk_scan(const float* __restrict__ Ag, const float* __restrict__ pw,
                   float* __restrict__ rbuf, float* __restrict__ Tbuf) {
  __shared__ float Msh[DS_ * LDST];
  __shared__ float Abuf[2][NN_];               // double-buffered A_t tiles
  __shared__ float wsh[DS_];
  int tid  = threadIdx.x;
  int lane = tid & 31;
  int wv   = tid >> 5;            // 0..15
  int ti = wv >> 2, tj = wv & 3;  // 4x4 tile grid of the 64x64 product
  int chunk = blockIdx.x;         // position = chunk*G_ + t
  size_t posbase = (size_t)chunk * G_;

  tile_load_async(&Abuf[0][0], Ag + posbase * NN_, tid);   // prefetch A_0

  for (int idx = tid; idx < NN_; idx += 512) {  // M = I (overlaps the async)
    int i = idx >> 6, j = idx & 63;
    Msh[i * LDST + j] = (i == j) ? 1.0f : 0.0f;
  }
  if (tid < DS_) wsh[tid] = pw[tid];
  __syncthreads();

  int hi = (lane >= 16) ? 1 : 0;
  int mr = lane & 15;
  for (int t = 0; t < G_; ++t) {
    int cur = t & 1, nxt = cur ^ 1;
    if (t + 1 < G_) {             // prefetch A_{t+1}; keep it in flight
      tile_load_async(&Abuf[nxt][0], Ag + (posbase + (size_t)(t + 1)) * NN_, tid);
      tile_wait<2>();             // wait only for A_t's two b128 ops
    } else {
      tile_wait<0>();
    }
    __syncthreads();              // every wave's slice of A_t present
    const float* At = &Abuf[cur][0];
    v8f c = {};
    #pragma unroll
    for (int kk = 0; kk < 16; ++kk) {          // K = 64 in steps of 4
      int k = kk * 4 + (hi ? 2 : 0);
      v2f a, b;
      const float* mp = &Msh[(ti * 16 + mr) * LDST + k];
      a.x = mp[0]; a.y = mp[1];                // M[row, k..k+1]
      b.x = At[(size_t)k       * DS_ + tj * 16 + mr];
      b.y = At[(size_t)(k + 1) * DS_ + tj * 16 + mr];
      c = __builtin_amdgcn_wmma_f32_16x16x4_f32(false, a, false, b,
                                                (short)0, c, false, false);
    }
    __syncthreads();                            // all reads of Msh/Abuf done
    int rb = ti * 16 + (hi ? 8 : 0);
    int cn = tj * 16 + mr;
    #pragma unroll
    for (int i = 0; i < 8; ++i) Msh[(rb + i) * LDST + cn] = c[i];
    __syncthreads();                            // new product visible
    if (tid < DS_) {                            // r_t = Lloc_t @ proj_w
      float acc = 0.0f;
      #pragma unroll 16
      for (int j = 0; j < DS_; ++j) acc += Msh[tid * LDST + j] * wsh[j];
      rbuf[(posbase + (size_t)t) * DS_ + tid] = acc;
    }
  }
  __syncthreads();
  for (int idx = tid; idx < NN_; idx += 512) {  // chunk total T_c
    int i = idx >> 6, j = idx & 63;
    Tbuf[(size_t)chunk * NN_ + idx] = Msh[i * LDST + j];
  }
}

// ------------- K3: exclusive scan of chunk totals per batch ----------------
__global__ __launch_bounds__(512)
void k3_scan_totals(const float* __restrict__ Tbuf, float* __restrict__ Ebuf) {
  __shared__ float Msh[DS_ * LDST];
  __shared__ float Tsh[2][NN_];
  int tid = threadIdx.x, lane = tid & 31, wv = tid >> 5;
  int ti = wv >> 2, tj = wv & 3;
  int b = blockIdx.x;

  tile_load_async(&Tsh[0][0], Tbuf + (size_t)(b * NC_) * NN_, tid);  // T_0

  for (int idx = tid; idx < NN_; idx += 512) {  // E_0 = I
    int i = idx >> 6, j = idx & 63;
    float v = (i == j) ? 1.0f : 0.0f;
    Msh[i * LDST + j] = v;
    Ebuf[(size_t)(b * NC_) * NN_ + idx] = v;
  }
  __syncthreads();
  int hi = (lane >= 16) ? 1 : 0;
  int mr = lane & 15;
  for (int cc = 1; cc < NC_; ++cc) {
    int cur = (cc - 1) & 1, nxt = cur ^ 1;
    if (cc + 1 < NC_) {
      tile_load_async(&Tsh[nxt][0], Tbuf + (size_t)(b * NC_ + cc) * NN_, tid);
      tile_wait<2>();
    } else {
      tile_wait<0>();
    }
    __syncthreads();
    const float* T = &Tsh[cur][0];
    v8f c = {};
    #pragma unroll
    for (int kk = 0; kk < 16; ++kk) {
      int k = kk * 4 + (hi ? 2 : 0);
      v2f a, bb;
      const float* mp = &Msh[(ti * 16 + mr) * LDST + k];
      a.x = mp[0]; a.y = mp[1];
      bb.x = T[(size_t)k       * DS_ + tj * 16 + mr];
      bb.y = T[(size_t)(k + 1) * DS_ + tj * 16 + mr];
      c = __builtin_amdgcn_wmma_f32_16x16x4_f32(false, a, false, bb,
                                                (short)0, c, false, false);
    }
    __syncthreads();
    int rb = ti * 16 + (hi ? 8 : 0);
    int cn = tj * 16 + mr;
    float* Eo = Ebuf + (size_t)(b * NC_ + cc) * NN_;
    #pragma unroll
    for (int i = 0; i < 8; ++i) {
      Msh[(rb + i) * LDST + cn] = c[i];
      Eo[(size_t)(rb + i) * DS_ + cn] = c[i];
    }
    __syncthreads();
  }
}

// --------------- K4: s = E_chunk @ r + proj_b (16 rows per wave) -----------
__global__ __launch_bounds__(256)
void k4_apply(const float* __restrict__ rbuf, const float* __restrict__ Ebuf,
              const float* __restrict__ pb, float* __restrict__ sbuf) {
  int gw   = (int)((blockIdx.x * blockDim.x + threadIdx.x) >> 5);  // 0..511
  int lane = threadIdx.x & 31;
  int hi = (lane >= 16) ? 1 : 0;
  int mr = lane & 15;
  size_t prow = (size_t)gw * 16;
  int b = (int)(prow / L_);
  int l = (int)(prow % L_);
  const float* E = Ebuf + (size_t)(b * NC_ + l / G_) * NN_;
  float pbv = pb[0];
  #pragma unroll
  for (int tjj = 0; tjj < 4; ++tjj) {           // N = 64 in 16-wide tiles
    v8f c = {};
    #pragma unroll
    for (int kk = 0; kk < 16; ++kk) {
      int k = kk * 4 + (hi ? 2 : 0);
      v2f a, bb;
      const float* rp = rbuf + (prow + mr) * DS_ + k;
      a.x = rp[0]; a.y = rp[1];
      bb.x = E[(size_t)(tjj * 16 + mr) * DS_ + k];      // B[k][n] = E[n][k]
      bb.y = E[(size_t)(tjj * 16 + mr) * DS_ + k + 1];
      c = __builtin_amdgcn_wmma_f32_16x16x4_f32(false, a, false, bb,
                                                (short)0, c, false, false);
    }
    int rb = hi ? 8 : 0;
    #pragma unroll
    for (int i = 0; i < 8; ++i)
      sbuf[(prow + rb + i) * DS_ + tjj * 16 + mr] = c[i] + pbv;
  }
}

// -------------- K5: out = s@C^T + x@D^T + Cb + Db (bf16 WMMA) --------------
__global__ __launch_bounds__(256)
void k5_out(const float* __restrict__ x, const float* __restrict__ sbuf,
            const float* __restrict__ Cw, const float* __restrict__ Cb,
            const float* __restrict__ Dw, const float* __restrict__ Db,
            float* __restrict__ out) {
  const int NTN = DM_ / 16;                     // 64 tiles in N
  int gw   = (int)((blockIdx.x * blockDim.x + threadIdx.x) >> 5);
  int lane = threadIdx.x & 31;
  int tm = gw / NTN, tn = gw % NTN;
  int hi = (lane >= 16) ? 1 : 0;
  int mr = lane & 15;
  size_t m = (size_t)tm * 16 + mr;
  size_t n = (size_t)tn * 16 + mr;
  v8f c = {};
  for (int k0 = 0; k0 < DM_; k0 += 32) {        // x @ D^T
    v16bf a, bf;
    const float* xr = x  + m * DM_ + k0 + (hi ? 16 : 0);
    const float* dp = Dw + n * DM_ + k0 + (hi ? 16 : 0);   // D^T[k][n]=Dw[n][k]
    __builtin_prefetch(xr + 32, 0, 1);
    #pragma unroll
    for (int i = 0; i < 16; ++i) a[i] = (__bf16)xr[i];
    #pragma unroll
    for (int i = 0; i < 16; ++i) bf[i] = (__bf16)dp[i];
    c = __builtin_amdgcn_wmma_f32_16x16x32_bf16(false, a, false, bf,
                                                (short)0, c, false, false);
  }
  for (int k0 = 0; k0 < DS_; k0 += 32) {        // s @ C^T
    v16bf a, bf;
    const float* sr = sbuf + m * DS_ + k0 + (hi ? 16 : 0);
    const float* cp = Cw   + n * DS_ + k0 + (hi ? 16 : 0); // C^T[k][n]=Cw[n][k]
    #pragma unroll
    for (int i = 0; i < 16; ++i) a[i] = (__bf16)sr[i];
    #pragma unroll
    for (int i = 0; i < 16; ++i) bf[i] = (__bf16)cp[i];
    c = __builtin_amdgcn_wmma_f32_16x16x32_bf16(false, a, false, bf,
                                                (short)0, c, false, false);
  }
  float bias = Cb[n] + Db[n];
  int rb = tm * 16 + (hi ? 8 : 0);
  #pragma unroll
  for (int i = 0; i < 8; ++i)
    out[(size_t)(rb + i) * DM_ + n] = c[i] + bias;
}

// ---------------------------------------------------------------------------
extern "C" void kernel_launch(void* const* d_in, const int* in_sizes, int n_in,
                              void* d_out, int out_size, void* d_ws, size_t ws_size,
                              hipStream_t stream) {
  const float* x  = (const float*)d_in[0];
  const float* A  = (const float*)d_in[1];
  const float* pw = (const float*)d_in[2];
  const float* pb = (const float*)d_in[3];
  const float* Cw = (const float*)d_in[4];
  const float* Cb = (const float*)d_in[5];
  const float* Dw = (const float*)d_in[6];
  const float* Db = (const float*)d_in[7];
  float* outp = (float*)d_out;

  // Workspace partition (floats): Ag 128MB | r 2MB | T 2MB | E 2MB | s 2MB
  float* wsf  = (float*)d_ws;
  float* Ag   = wsf;
  float* rbuf = Ag   + (size_t)BL_ * NN_;
  float* Tbuf = rbuf + (size_t)BL_ * DS_;
  float* Ebuf = Tbuf + (size_t)B_ * NC_ * NN_;
  float* sbuf = Ebuf + (size_t)B_ * NC_ * NN_;

  k1_xA_gelu    <<<16384, 256, 0, stream>>>(x, A, Ag);
  k2_chunk_scan <<<B_ * NC_, 512, 0, stream>>>(Ag, pw, rbuf, Tbuf);
  k3_scan_totals<<<B_, 512, 0, stream>>>(Tbuf, Ebuf);
  k4_apply      <<<64, 256, 0, stream>>>(rbuf, Ebuf, pb, sbuf);
  k5_out        <<<4096, 256, 0, stream>>>(x, sbuf, Cw, Cb, Dw, Db, outp);
}
